// SP_Memory_Net_v1_68410239090766
// MI455X (gfx1250) — matrix operand order repaired and checked by reading, not compiled, not run on hardware
//
#include <hip/hip_runtime.h>
#include <hip/hip_bf16.h>

typedef __attribute__((ext_vector_type(16))) __bf16 v16bf;
typedef __attribute__((ext_vector_type(8)))  float  v8f;
typedef __attribute__((ext_vector_type(4)))  float  v4f;

#define POSE 512
#define CHUNK 32
#define PRED 128
#define BATCH 2048

// Convert 16 contiguous f32 at p -> v16bf (emits v_cvt_pk_bf16_f32 pairs).
__device__ __forceinline__ v16bf cvt16(const float* p)
{
    v4f t0 = *(const v4f*)(p);
    v4f t1 = *(const v4f*)(p + 4);
    v4f t2 = *(const v4f*)(p + 8);
    v4f t3 = *(const v4f*)(p + 12);
    v16bf r;
    r[0]  = (__bf16)t0.x; r[1]  = (__bf16)t0.y; r[2]  = (__bf16)t0.z; r[3]  = (__bf16)t0.w;
    r[4]  = (__bf16)t1.x; r[5]  = (__bf16)t1.y; r[6]  = (__bf16)t1.z; r[7]  = (__bf16)t1.w;
    r[8]  = (__bf16)t2.x; r[9]  = (__bf16)t2.y; r[10] = (__bf16)t2.z; r[11] = (__bf16)t2.w;
    r[12] = (__bf16)t3.x; r[13] = (__bf16)t3.y; r[14] = (__bf16)t3.z; r[15] = (__bf16)t3.w;
    return r;
}

// A fragment: elems 0..7 from p0 (8 f32), elems 8..15 from p1 (8 f32).
__device__ __forceinline__ v16bf cvtA(const float* p0, const float* p1)
{
    v4f t0 = *(const v4f*)(p0);
    v4f t1 = *(const v4f*)(p0 + 4);
    v4f t2 = *(const v4f*)(p1);
    v4f t3 = *(const v4f*)(p1 + 4);
    v16bf r;
    r[0]  = (__bf16)t0.x; r[1]  = (__bf16)t0.y; r[2]  = (__bf16)t0.z; r[3]  = (__bf16)t0.w;
    r[4]  = (__bf16)t1.x; r[5]  = (__bf16)t1.y; r[6]  = (__bf16)t1.z; r[7]  = (__bf16)t1.w;
    r[8]  = (__bf16)t2.x; r[9]  = (__bf16)t2.y; r[10] = (__bf16)t2.z; r[11] = (__bf16)t2.w;
    r[12] = (__bf16)t3.x; r[13] = (__bf16)t3.y; r[14] = (__bf16)t3.z; r[15] = (__bf16)t3.w;
    return r;
}

// ---------------------------------------------------------------------------
// GEMM:  Out[m, n] = sum_k A[aOff + m*aStride + k] * Bm[n*K + k] + bias[n]
// M = 2048, N = 512.  bf16 WMMA 16x16x32, f32 accumulate.
//
// Block = 256 threads = 8 waves, arranged 4 (M) x 2 (N).
// Block tile = 128M x 128N; wave tile = 32M x 64N = 2x4 WMMA tiles.
// B tile (128 rows x 32 K, f32, 16 KB) is staged into LDS per K-step with
// GLOBAL_LOAD_ASYNC_TO_LDS_B128 (ASYNCcnt), shared by all 8 waves.
//
// VGPR layouts used here (wave32; A/B K order is contiguous, not the ISA table):
//   A 16x32 bf16 : lane l (h=l>>4, m=l&15): elem  e     -> K = 16h + e
//   B 32x16 bf16 : lane l (h=l>>4, n=l&15): elem  e     -> K = 16h + e
//   C 16x16 f32  : lane l, reg r           -> M = r + 8h, N = l&15
// ---------------------------------------------------------------------------
__global__ __launch_bounds__(256)
void gemm_bf16_wmma(const float* __restrict__ A, long aOff, long aStride,
                    const float* __restrict__ Bm,
                    const float* __restrict__ bias,
                    float* __restrict__ Out, int K)
{
    __shared__ float sB[128 * 32];            // B tile: [n(128)][k(32)] f32

    const int tid  = threadIdx.x;
    const int lane = tid & 31;
    const int wave = tid >> 5;                // 0..7
    const int wm   = wave & 3;                // M sub-tile
    const int wn   = wave >> 2;               // N sub-tile (0..1)

    const int mBase = blockIdx.x * 128 + wm * 32;
    const int nBlk  = blockIdx.y * 128;       // block's N origin
    const int nLoc  = wn * 64;                // wave's N origin inside tile

    const int h  = lane >> 4;
    const int ml = lane & 15;
    const int nl = lane & 15;

    const float* aRow0 = A + aOff + (long)(mBase +      ml) * aStride;
    const float* aRow1 = A + aOff + (long)(mBase + 16 + ml) * aStride;

    // Async staging plan: 1024 x 16B chunks, 4 per thread.
    // chunk c = tid + 256*i : row = c >> 3, kOff(floats) = (c & 7) * 4
    const float* gStage[4];
    unsigned     ldsOff[4];
    #pragma unroll
    for (int i = 0; i < 4; ++i) {
        const int c    = tid + 256 * i;
        const int row  = c >> 3;
        const int kOff = (c & 7) * 4;
        gStage[i] = Bm + (long)(nBlk + row) * K + kOff;
        ldsOff[i] = (unsigned)(size_t)(&sB[row * 32 + kOff]);
    }

    v8f c0[4] = {{}, {}, {}, {}};             // rows mBase + 0..15
    v8f c1[4] = {{}, {}, {}, {}};             // rows mBase + 16..31

    for (int k0 = 0; k0 < K; k0 += 32) {
        __syncthreads();                       // sB reads from prev step done

        // ---- issue async copy of B tile into LDS (ASYNCcnt) ----
        #pragma unroll
        for (int i = 0; i < 4; ++i) {
            unsigned long long ga = (unsigned long long)(size_t)(gStage[i] + k0);
            asm volatile("global_load_async_to_lds_b128 %0, %1, off"
                         :: "v"(ldsOff[i]), "v"(ga) : "memory");
        }

        // ---- A fragments straight from global while async is in flight ----
        v16bf a0 = cvtA(aRow0 + k0 + 16 * h, aRow0 + k0 + 16 * h + 8);
        v16bf a1 = cvtA(aRow1 + k0 + 16 * h, aRow1 + k0 + 16 * h + 8);

        asm volatile("s_wait_asynccnt 0" ::: "memory");
        __syncthreads();                       // B tile visible to all waves

        // ---- B fragments from LDS, 8 WMMAs ----
        #pragma unroll
        for (int t = 0; t < 4; ++t) {
            const float* p = sB + (nLoc + 16 * t + nl) * 32 + 16 * h;
            v16bf b = cvt16(p);
            c0[t] = __builtin_amdgcn_wmma_f32_16x16x32_bf16(false, a0, false, b,
                                                            (short)0, c0[t], false, false);
            c1[t] = __builtin_amdgcn_wmma_f32_16x16x32_bf16(false, a1, false, b,
                                                            (short)0, c1[t], false, false);
        }
    }

    // ---- epilogue: add bias, store f32 ----
    #pragma unroll
    for (int t = 0; t < 4; ++t) {
        const int  n  = nBlk + nLoc + 16 * t + nl;
        const float bv = bias[n];
        #pragma unroll
        for (int r = 0; r < 8; ++r) {
            const long m0 = mBase +      r + 8 * h;
            const long m1 = mBase + 16 + r + 8 * h;
            Out[m0 * (long)POSE + n] = c0[t][r] + bv;
            Out[m1 * (long)POSE + n] = c1[t][r] + bv;
        }
    }
}

// ---------------------------------------------------------------------------
// Epilogue: score -> sigmoid gate -> blend first CHUNK rows, copy the rest.
// One block per batch row.
// ---------------------------------------------------------------------------
__global__ __launch_bounds__(256)
void gate_blend_kernel(const float* __restrict__ pred,
                       const float* __restrict__ mem,
                       float* __restrict__ out)
{
    const int b = blockIdx.x;
    const float* memRow = mem  + (long)b * POSE;
    const float* predB  = pred + (long)b * PRED * POSE;
    float*       outB   = out  + (long)b * PRED * POSE;

    __shared__ float sMem[POSE];
    __shared__ float sG[CHUNK];

    const int tid = threadIdx.x;
    for (int i = tid; i < POSE; i += 256) sMem[i] = memRow[i];
    __syncthreads();

    // 32 score groups x 8 lanes; each group lives inside one wave32.
    const int grp = tid >> 3;
    const int lag = tid & 7;
    {
        const float* headRow = predB + (long)grp * POSE;
        float s = 0.f;
        for (int d = lag; d < POSE; d += 8)
            s += sMem[d] * headRow[d];
        s += __shfl_xor(s, 1, 32);
        s += __shfl_xor(s, 2, 32);
        s += __shfl_xor(s, 4, 32);
        if (lag == 0) sG[grp] = 1.0f / (1.0f + __expf(-s));
    }
    __syncthreads();

    const int headV4 = CHUNK * POSE / 4;      // 4096
    for (int i = tid; i < headV4; i += 256) {
        const int c     = i >> 7;             // 128 v4 per row
        const int dBase = (i & 127) * 4;
        const float g   = sG[c];
        v4f hd = ((const v4f*)predB)[i];
        v4f mm = *(const v4f*)(sMem + dBase);
        ((v4f*)outB)[i] = g * hd + (1.0f - g) * mm;
    }
    const int totV4 = PRED * POSE / 4;        // 16384
    for (int i = tid + headV4; i < totV4; i += 256) {
        ((v4f*)outB)[i] = ((const v4f*)predB)[i];
    }
}

extern "C" void kernel_launch(void* const* d_in, const int* in_sizes, int n_in,
                              void* d_out, int out_size, void* d_ws, size_t ws_size,
                              hipStream_t stream)
{
    const float* IF = (const float*)d_in[0];   // (B, 64, 512)
    const float* PF = (const float*)d_in[1];   // (B, 128, 512)
    const float* W1 = (const float*)d_in[2];   // (512, 16384)
    const float* b1 = (const float*)d_in[3];   // (512,)
    const float* W2 = (const float*)d_in[4];   // (512, 512)
    const float* b2 = (const float*)d_in[5];   // (512,)
    float* out = (float*)d_out;

    float* h   = (float*)d_ws;                 // 2048*512 f32 = 4 MB
    float* mem = h + (long)BATCH * POSE;       // 2048*512 f32 = 4 MB

    dim3 blk(256);
    dim3 grid(BATCH / 128, POSE / 128);        // 16 x 4

    // GEMM1: x = initial_feature[:, 32:64, :] -> contiguous 16384-float run
    // per batch row at offset 16384, stride 32768.
    gemm_bf16_wmma<<<grid, blk, 0, stream>>>(IF, (long)(CHUNK * POSE),
                                             (long)(2 * CHUNK * POSE),
                                             W1, b1, h, CHUNK * POSE);
    // GEMM2: mem = h @ W2^T + b2
    gemm_bf16_wmma<<<grid, blk, 0, stream>>>(h, 0L, (long)POSE,
                                             W2, b2, mem, POSE);
    // Gate + blend + pass-through copy
    gate_blend_kernel<<<BATCH, blk, 0, stream>>>(PF, mem, out);
}
